// UnimolPlusEncoderLayer_67611375173980
// MI455X (gfx1250) — matrix operand order repaired and checked by reading, not compiled, not run on hardware
//
// CDNA5 (gfx1250) implementation of the UnimolPlus encoder layer.
// All large GEMMs use v_wmma_f32_16x16x32_bf16 (wave32, 16x16 tiles, f32 acc).
// Pair-tensor row ops (P=64) are fused single-pass VALU kernels (memory-bound:
// ~1.3GB of pair/tri traffic vs ~75 GFLOP total -> HBM-limited at 23.3 TB/s).
// GEMM staging uses float4 vector loads + packed bf16 b64 LDS stores on the
// (common) full-tile path; guarded scalar staging only for partial tiles.
//
// Input order (setup_inputs dict order, params flattened in insertion order):
//  0 x (B,N,D) 1 pair (B,N,N,P) 2 pair_mask (B,N,N) 3 self_attn_mask (B,1,1,N)
//  4 op_mask (B,N,1) 5 op_norm (B,1,1,1) then 39 params (see code).
// Output: concat[x (B,N,D), pair (B,N,N,P)] as f32.

#include <hip/hip_runtime.h>
#include <math.h>

#define DEVFN static __device__ __forceinline__

typedef __attribute__((ext_vector_type(16))) __bf16 v16bf;
typedef __attribute__((ext_vector_type(8)))  float  v8f;

static constexpr int  Bb  = 2;
static constexpr int  Nn  = 512;
static constexpr int  Dd  = 768;
static constexpr int  Pp  = 64;
static constexpr int  HIDh= 32;
static constexpr int  FFf = 3072;
static constexpr int  Hh  = 8;
static constexpr int  HDd = 96;
static constexpr long long NNll = (long long)Nn * Nn;

// ---------------- helpers ----------------

DEVFN unsigned short f2bf(float f) {
  union { float f; unsigned int u; } x; x.f = f;
  unsigned int u = x.u;
  if ((u & 0x7f800000u) == 0x7f800000u) return (unsigned short)(u >> 16); // inf/nan
  unsigned int r = u + 0x7fffu + ((u >> 16) & 1u);                        // RNE
  return (unsigned short)(r >> 16);
}

DEVFN unsigned int f2bf2(float a, float b) {
  return (unsigned int)f2bf(a) | ((unsigned int)f2bf(b) << 16);
}

DEVFN float gelu_exact(float x) {
  return 0.5f * x * (1.0f + erff(x * 0.70710678118654752440f));
}

DEVFN void ln64(float* v, const float* __restrict__ g, const float* __restrict__ b) {
  float mu = 0.f;
#pragma unroll
  for (int i = 0; i < 64; ++i) mu += v[i];
  mu *= (1.0f / 64.0f);
  float var = 0.f;
#pragma unroll
  for (int i = 0; i < 64; ++i) { float d = v[i] - mu; var += d * d; }
  float r = rsqrtf(var * (1.0f / 64.0f) + 1e-5f);
#pragma unroll
  for (int i = 0; i < 64; ++i) v[i] = (v[i] - mu) * r * g[i] + b[i];
}

// A fragment: 16x32 bf16, M = lane&15, half = lane>>4.
// VGPR i<4: K = 2i + 8*half ; i>=4: K = 16 + 2(i-4) + 8*half (ISA 7.12.2).
DEVFN v16bf load_a_frag(const unsigned short* rowPtr, int half) {
  union { v16bf v; unsigned int u[8]; } r;
  const unsigned int* p = (const unsigned int*)rowPtr;
#pragma unroll
  for (int i = 0; i < 8; ++i) {
    int kbase = ((i < 4) ? (2 * i) : (16 + 2 * (i - 4))) + 8 * half;
    r.u[i] = p[kbase >> 1];
  }
  return r.v;
}

// B fragment: 32x16 bf16, N = lane&15, same K order as A: VGPR i<4: K = 2i + 8*half ; i>=4: K = 16 + 2(i-4) + 8*half.
// Bs is stored [n][k] (K contiguous per column).
DEVFN v16bf load_b_frag(const unsigned short* colPtr, int half) {
  union { v16bf v; unsigned int u[8]; } r;
  const unsigned int* p = (const unsigned int*)colPtr;
#pragma unroll
  for (int i = 0; i < 8; ++i) r.u[i] = p[((i < 4) ? i : (8 + (i - 4))) + 4 * half];
  return r.v;
}

// ---------------- generic batched WMMA GEMM ----------------
// C[m,n] = epi( alpha * sum_k Aop[m,k]*Bop[k,n] + bias[n] ) (*rowScale[m]) (+C)
// Aop = TA ? A^T (A stored KxM) : A (MxK);  Bop = TB ? B^T (B stored NxK) : B.
// All call sites have K % 32 == 0 and 16-byte aligned rows (lda/ldb % 4 == 0).

struct GemmArgs {
  const float* A; const float* B; float* C;
  const float* bias;        // per-column, nullable
  const float* rowScale;    // per-row, nullable
  long long rs1, rs2;
  int M, N, K, lda, ldb, ldc;
  long long sA1, sA2, sB1, sB2, sC1, sC2;
  int inner;                // batch z -> (z/inner, z%inner)
  int batch;                // grid.z
  float alpha;
  int flags;                // bit0 = gelu, bit1 = accumulate into C
};

template <int TA, int TB>
__global__ __launch_bounds__(128) void gemm_wmma(GemmArgs g) {
  __shared__ unsigned short As[64 * 40];
  __shared__ unsigned short Bs[64 * 40];
  const int tid  = threadIdx.x;
  const int lane = tid & 31, wv = tid >> 5;
  const int wr = (wv >> 1) * 32, wc = (wv & 1) * 32;
  const int half = lane >> 4, l16 = lane & 15;
  const long long z1 = blockIdx.z / g.inner, z2 = blockIdx.z % g.inner;
  const float* A  = g.A + z1 * g.sA1 + z2 * g.sA2;
  const float* Bm = g.B + z1 * g.sB1 + z2 * g.sB2;
  float*       C  = g.C + z1 * g.sC1 + z2 * g.sC2;
  const int m0 = blockIdx.y * 64, n0 = blockIdx.x * 64;
  const bool full = (m0 + 64 <= g.M) && (n0 + 64 <= g.N);

  v8f acc[2][2];
#pragma unroll
  for (int a = 0; a < 2; ++a)
#pragma unroll
    for (int b = 0; b < 2; ++b)
#pragma unroll
      for (int e = 0; e < 8; ++e) acc[a][b][e] = 0.0f;

  for (int k0 = 0; k0 < g.K; k0 += 32) {
    if (full) {
      // -------- fast path: float4 vector loads, packed bf16 stores --------
#pragma unroll
      for (int i = 0; i < 4; ++i) {
        int vi = tid + i * 128;                       // 512 float4s = 64x32
        if (TA == 0) {
          int m = vi >> 3, k = (vi & 7) << 2;
          float4 f = *(const float4*)(A + (long long)(m0 + m) * g.lda + (k0 + k));
          *(uint2*)(&As[m * 40 + k]) = make_uint2(f2bf2(f.x, f.y), f2bf2(f.z, f.w));
        } else {
          int k = vi >> 4, m = (vi & 15) << 2;
          float4 f = *(const float4*)(A + (long long)(k0 + k) * g.lda + (m0 + m));
          As[(m + 0) * 40 + k] = f2bf(f.x);
          As[(m + 1) * 40 + k] = f2bf(f.y);
          As[(m + 2) * 40 + k] = f2bf(f.z);
          As[(m + 3) * 40 + k] = f2bf(f.w);
        }
      }
#pragma unroll
      for (int i = 0; i < 4; ++i) {
        int vi = tid + i * 128;
        if (TB == 1) {
          int n = vi >> 3, k = (vi & 7) << 2;
          float4 f = *(const float4*)(Bm + (long long)(n0 + n) * g.ldb + (k0 + k));
          *(uint2*)(&Bs[n * 40 + k]) = make_uint2(f2bf2(f.x, f.y), f2bf2(f.z, f.w));
        } else {
          int k = vi >> 4, n = (vi & 15) << 2;
          float4 f = *(const float4*)(Bm + (long long)(k0 + k) * g.ldb + (n0 + n));
          Bs[(n + 0) * 40 + k] = f2bf(f.x);
          Bs[(n + 1) * 40 + k] = f2bf(f.y);
          Bs[(n + 2) * 40 + k] = f2bf(f.z);
          Bs[(n + 3) * 40 + k] = f2bf(f.w);
        }
      }
    } else {
      // -------- guarded scalar path (partial tiles only) --------
#pragma unroll 4
      for (int idx = tid; idx < 64 * 32; idx += 128) {
        int m = idx >> 5, kk = idx & 31;
        int gm = m0 + m, gk = k0 + kk;
        float v = 0.0f;
        if (gm < g.M && gk < g.K)
          v = TA ? A[(long long)gk * g.lda + gm] : A[(long long)gm * g.lda + gk];
        As[m * 40 + kk] = f2bf(v);
      }
#pragma unroll 4
      for (int idx = tid; idx < 64 * 32; idx += 128) {
        int n = idx >> 5, kk = idx & 31;
        int gn = n0 + n, gk = k0 + kk;
        float v = 0.0f;
        if (gn < g.N && gk < g.K)
          v = TB ? Bm[(long long)gn * g.ldb + gk] : Bm[(long long)gk * g.ldb + gn];
        Bs[n * 40 + kk] = f2bf(v);
      }
    }
    __syncthreads();
    v16bf af0 = load_a_frag(&As[(wr + l16) * 40], half);
    v16bf af1 = load_a_frag(&As[(wr + 16 + l16) * 40], half);
    v16bf bf0 = load_b_frag(&Bs[(wc + l16) * 40], half);
    v16bf bf1 = load_b_frag(&Bs[(wc + 16 + l16) * 40], half);
    acc[0][0] = __builtin_amdgcn_wmma_f32_16x16x32_bf16(false, af0, false, bf0, (short)0, acc[0][0], false, false);
    acc[0][1] = __builtin_amdgcn_wmma_f32_16x16x32_bf16(false, af0, false, bf1, (short)0, acc[0][1], false, false);
    acc[1][0] = __builtin_amdgcn_wmma_f32_16x16x32_bf16(false, af1, false, bf0, (short)0, acc[1][0], false, false);
    acc[1][1] = __builtin_amdgcn_wmma_f32_16x16x32_bf16(false, af1, false, bf1, (short)0, acc[1][1], false, false);
    __syncthreads();
  }

  // C layout (ISA 7.12.2): VGPR e -> row = 8*half + e, col = lane&15.
#pragma unroll
  for (int mi = 0; mi < 2; ++mi)
#pragma unroll
    for (int ni = 0; ni < 2; ++ni)
#pragma unroll
      for (int e = 0; e < 8; ++e) {
        int m = m0 + wr + mi * 16 + half * 8 + e;
        int n = n0 + wc + ni * 16 + l16;
        if (m < g.M && n < g.N) {
          float val = acc[mi][ni][e] * g.alpha;
          if (g.bias)     val += g.bias[n];
          if (g.rowScale) val *= g.rowScale[z1 * g.rs1 + z2 * g.rs2 + m];
          if (g.flags & 1) val = gelu_exact(val);
          float* cp = C + (long long)m * g.ldc + n;
          if (g.flags & 2) val += *cp;
          *cp = val;
        }
      }
}

// ---------------- attention pair-bias + mask ----------------
// scores[b,h,i,j] += sum_p pair[b,i,j,p]*Wbias[p,h] + bbias[h] + mask[b,j]
__global__ __launch_bounds__(256) void bias_mask_kernel(
    const float* __restrict__ pair, const float* __restrict__ Wbias,
    const float* __restrict__ bbias, const float* __restrict__ mask,
    float* __restrict__ scores) {
  __shared__ float sW[64 * 8];
  __shared__ float sb[8];
  int tid = threadIdx.x;
  for (int i = tid; i < 512; i += 256) sW[i] = Wbias[i];
  if (tid < 8) sb[tid] = bbias[tid];
  __syncthreads();
  long long gid = (long long)blockIdx.x * 256 + tid;   // total B*N*N
  int j = (int)(gid % Nn);
  long long t = gid / Nn;
  int i = (int)(t % Nn);
  int b = (int)(t / Nn);
  float s[8];
#pragma unroll
  for (int h = 0; h < 8; ++h) s[h] = 0.f;
  const float4* pr = (const float4*)(pair + gid * 64);
#pragma unroll
  for (int q = 0; q < 16; ++q) {
    float4 v = pr[q];
#pragma unroll
    for (int h = 0; h < 8; ++h)
      s[h] += v.x * sW[(q * 4 + 0) * 8 + h] + v.y * sW[(q * 4 + 1) * 8 + h] +
              v.z * sW[(q * 4 + 2) * 8 + h] + v.w * sW[(q * 4 + 3) * 8 + h];
  }
  float mk = mask[(long long)b * Nn + j];
#pragma unroll
  for (int h = 0; h < 8; ++h) {
    long long idx = ((((long long)b * Hh + h) * Nn + i) * Nn) + j;
    scores[idx] += s[h] + sb[h] + mk;
  }
}

// ---------------- row softmax over N=512 ----------------
__global__ __launch_bounds__(256) void softmax_kernel(float* __restrict__ scores) {
  __shared__ float red[256];
  int tid = threadIdx.x;
  float* p = scores + (long long)blockIdx.x * Nn;
  float v0 = p[tid], v1 = p[tid + 256];
  red[tid] = fmaxf(v0, v1);
  __syncthreads();
  for (int s = 128; s > 0; s >>= 1) { if (tid < s) red[tid] = fmaxf(red[tid], red[tid + s]); __syncthreads(); }
  float mx = red[0];
  __syncthreads();
  float e0 = __expf(v0 - mx), e1 = __expf(v1 - mx);
  red[tid] = e0 + e1;
  __syncthreads();
  for (int s = 128; s > 0; s >>= 1) { if (tid < s) red[tid] += red[tid + s]; __syncthreads(); }
  float inv = 1.0f / red[0];
  p[tid] = e0 * inv;
  p[tid + 256] = e1 * inv;
}

// ---------------- residual + LayerNorm over D=768 ----------------
__global__ __launch_bounds__(256) void ln_residual_kernel(
    const float* __restrict__ xin, const float* __restrict__ delta,
    const float* __restrict__ gamma, const float* __restrict__ beta,
    float* __restrict__ out) {
  __shared__ float red[256];
  int tid = threadIdx.x;
  long long base = (long long)blockIdx.x * Dd;
  float v[3];
#pragma unroll
  for (int r = 0; r < 3; ++r) { int d = tid + r * 256; v[r] = xin[base + d] + delta[base + d]; }
  red[tid] = v[0] + v[1] + v[2];
  __syncthreads();
  for (int s = 128; s > 0; s >>= 1) { if (tid < s) red[tid] += red[tid + s]; __syncthreads(); }
  float mu = red[0] * (1.0f / Dd);
  __syncthreads();
  float q = 0.f;
#pragma unroll
  for (int r = 0; r < 3; ++r) { float d = v[r] - mu; q += d * d; }
  red[tid] = q;
  __syncthreads();
  for (int s = 128; s > 0; s >>= 1) { if (tid < s) red[tid] += red[tid + s]; __syncthreads(); }
  float rs = rsqrtf(red[0] * (1.0f / Dd) + 1e-5f);
#pragma unroll
  for (int r = 0; r < 3; ++r) {
    int d = tid + r * 256;
    out[base + d] = (v[r] - mu) * rs * gamma[d] + beta[d];
  }
}

// ---------------- OPM: pair = LN(pair + (z+bias_in_z)*op_norm) ----------------
__global__ __launch_bounds__(256) void opm_ln_kernel(
    const float* __restrict__ pairIn, const float* __restrict__ z,
    const float* __restrict__ opnorm, const float* __restrict__ gamma,
    const float* __restrict__ beta, float* __restrict__ out) {
  long long gid = (long long)blockIdx.x * 256 + threadIdx.x;
  int b = (int)(gid / NNll);
  float s = opnorm[b];
  float v[64];
  const float4* pi = (const float4*)(pairIn + gid * 64);
  const float4* zi = (const float4*)(z + gid * 64);
#pragma unroll
  for (int q = 0; q < 16; ++q) {
    float4 a = pi[q], c = zi[q];
    v[q * 4 + 0] = a.x + c.x * s;
    v[q * 4 + 1] = a.y + c.y * s;
    v[q * 4 + 2] = a.z + c.z * s;
    v[q * 4 + 3] = a.w + c.w * s;
  }
  ln64(v, gamma, beta);
  float4* o = (float4*)(out + gid * 64);
#pragma unroll
  for (int q = 0; q < 16; ++q)
    o[q] = make_float4(v[q * 4 + 0], v[q * 4 + 1], v[q * 4 + 2], v[q * 4 + 3]);
}

// ---------------- triangle projection: ta/tb channel-major ----------------
// v = (pair@Wabp + babp) * m * sigmoid(pair@Wabg + babg), m = mask * N^-1/2
__global__ __launch_bounds__(128) void tri_proj_kernel(
    const float* __restrict__ pairo, const float* __restrict__ pmask,
    const float* __restrict__ Wabp, const float* __restrict__ babp,
    const float* __restrict__ Wabg, const float* __restrict__ babg,
    float* __restrict__ ta, float* __restrict__ tb) {
  __shared__ float sWp[4096], sWg[4096], sbp[64], sbg[64];
  int tid = threadIdx.x;
  for (int i = tid; i < 4096; i += 128) { sWp[i] = Wabp[i]; sWg[i] = Wabg[i]; }
  if (tid < 64) { sbp[tid] = babp[tid]; sbg[tid] = babg[tid]; }
  __syncthreads();
  long long gid = (long long)blockIdx.x * 128 + tid;
  float row[64];
  const float4* pr = (const float4*)(pairo + gid * 64);
#pragma unroll
  for (int q = 0; q < 16; ++q) {
    float4 a = pr[q];
    row[q * 4 + 0] = a.x; row[q * 4 + 1] = a.y; row[q * 4 + 2] = a.z; row[q * 4 + 3] = a.w;
  }
  float m = pmask[gid] * 0.044194173824159216f;   // 512^-0.5
  long long b = gid / NNll, ik = gid % NNll;
#pragma unroll
  for (int p = 0; p < 64; ++p) {
    float sa = sbp[p], sg = sbg[p];
#pragma unroll
    for (int d = 0; d < 64; ++d) { sa += row[d] * sWp[d * 64 + p]; sg += row[d] * sWg[d * 64 + p]; }
    float vv = sa * m * (1.0f / (1.0f + __expf(-sg)));
    float* dst = (p < 32) ? ta : tb;
    // single-use channel-major stream: keep it out of the way of pairo in L2
    __builtin_nontemporal_store(vv, &dst[(b * 32 + (p & 31)) * NNll + ik]);
  }
}

// ---------------- triangle output: pair = LN(pair + g*(LN32(t)@Wz + bz)) ----
__global__ __launch_bounds__(128) void tri_out_kernel(
    const float* __restrict__ ttri, const float* __restrict__ pairo,
    const float* __restrict__ lnOg, const float* __restrict__ lnOb,
    const float* __restrict__ Wg, const float* __restrict__ bg,
    const float* __restrict__ Wz, const float* __restrict__ bz,
    const float* __restrict__ lnTg, const float* __restrict__ lnTb,
    float* __restrict__ out) {
  __shared__ float sWg[4096], sWz[2048], sbg[64], sbz[64], sg32[32], sb32[32];
  int tid = threadIdx.x;
  for (int i = tid; i < 4096; i += 128) sWg[i] = Wg[i];
  for (int i = tid; i < 2048; i += 128) sWz[i] = Wz[i];
  if (tid < 64) { sbg[tid] = bg[tid]; sbz[tid] = bz[tid]; }
  if (tid < 32) { sg32[tid] = lnOg[tid]; sb32[tid] = lnOb[tid]; }
  __syncthreads();
  long long gid = (long long)blockIdx.x * 128 + tid;
  long long b = gid / NNll, ik = gid % NNll;
  float tc[32];
#pragma unroll
  for (int c = 0; c < 32; ++c)
    tc[c] = __builtin_nontemporal_load(&ttri[(b * 32 + c) * NNll + ik]);
  float mu = 0.f;
#pragma unroll
  for (int c = 0; c < 32; ++c) mu += tc[c];
  mu *= (1.0f / 32.0f);
  float var = 0.f;
#pragma unroll
  for (int c = 0; c < 32; ++c) { float d = tc[c] - mu; var += d * d; }
  float rr = rsqrtf(var * (1.0f / 32.0f) + 1e-5f);
#pragma unroll
  for (int c = 0; c < 32; ++c) tc[c] = (tc[c] - mu) * rr * sg32[c] + sb32[c];

  float row[64];
  const float4* pr = (const float4*)(pairo + gid * 64);
#pragma unroll
  for (int q = 0; q < 16; ++q) {
    float4 a = pr[q];
    row[q * 4 + 0] = a.x; row[q * 4 + 1] = a.y; row[q * 4 + 2] = a.z; row[q * 4 + 3] = a.w;
  }
  float val[64];
#pragma unroll
  for (int p = 0; p < 64; ++p) {
    float gp = sbg[p];
#pragma unroll
    for (int d = 0; d < 64; ++d) gp += row[d] * sWg[d * 64 + p];
    float tz = sbz[p];
#pragma unroll
    for (int c = 0; c < 32; ++c) tz += tc[c] * sWz[c * 64 + p];
    val[p] = row[p] + gp * tz;
  }
  ln64(val, lnTg, lnTb);
  float4* o = (float4*)(out + gid * 64);
#pragma unroll
  for (int q = 0; q < 16; ++q)
    o[q] = make_float4(val[q * 4 + 0], val[q * 4 + 1], val[q * 4 + 2], val[q * 4 + 3]);
}

// ---------------- pair FFN: pair = LN(pair + gelu(pair@W1+b1)@W2+b2) -------
__global__ __launch_bounds__(128) void pair_ffn_kernel(
    float* __restrict__ pairio,
    const float* __restrict__ W1, const float* __restrict__ b1,
    const float* __restrict__ W2, const float* __restrict__ b2,
    const float* __restrict__ lng, const float* __restrict__ lnb) {
  __shared__ float sW1[4096], sW2[4096], sb1[64], sb2[64];
  int tid = threadIdx.x;
  for (int i = tid; i < 4096; i += 128) { sW1[i] = W1[i]; sW2[i] = W2[i]; }
  if (tid < 64) { sb1[tid] = b1[tid]; sb2[tid] = b2[tid]; }
  __syncthreads();
  long long gid = (long long)blockIdx.x * 128 + tid;
  float row[64];
  const float4* pr = (const float4*)(pairio + gid * 64);
#pragma unroll
  for (int q = 0; q < 16; ++q) {
    float4 a = pr[q];
    row[q * 4 + 0] = a.x; row[q * 4 + 1] = a.y; row[q * 4 + 2] = a.z; row[q * 4 + 3] = a.w;
  }
  float h[64];
#pragma unroll
  for (int f = 0; f < 64; ++f) {
    float s = sb1[f];
#pragma unroll
    for (int d = 0; d < 64; ++d) s += row[d] * sW1[d * 64 + f];
    h[f] = gelu_exact(s);
  }
#pragma unroll
  for (int p = 0; p < 64; ++p) {
    float s = sb2[p];
#pragma unroll
    for (int f = 0; f < 64; ++f) s += h[f] * sW2[f * 64 + p];
    row[p] += s;
  }
  ln64(row, lng, lnb);
  float4* o = (float4*)(pairio + gid * 64);
#pragma unroll
  for (int q = 0; q < 16; ++q)
    o[q] = make_float4(row[q * 4 + 0], row[q * 4 + 1], row[q * 4 + 2], row[q * 4 + 3]);
}

// ---------------- host side ----------------

static GemmArgs ga(const float* A, int lda, long long sA1, long long sA2,
                   const float* B, int ldb, long long sB1, long long sB2,
                   float* C, int ldc, long long sC1, long long sC2,
                   int M, int N, int K, int inner, int batch,
                   const float* bias, const float* rowScale,
                   float alpha, int flags) {
  GemmArgs g;
  g.A = A; g.B = B; g.C = C; g.bias = bias; g.rowScale = rowScale;
  g.rs1 = 0; g.rs2 = 0;
  g.M = M; g.N = N; g.K = K; g.lda = lda; g.ldb = ldb; g.ldc = ldc;
  g.sA1 = sA1; g.sA2 = sA2; g.sB1 = sB1; g.sB2 = sB2; g.sC1 = sC1; g.sC2 = sC2;
  g.inner = inner; g.batch = batch; g.alpha = alpha; g.flags = flags;
  return g;
}

static void launch_gemm(int TA, int TB, const GemmArgs& g, hipStream_t s) {
  dim3 grid((g.N + 63) / 64, (g.M + 63) / 64, g.batch);
  dim3 blk(128);
  if (!TA && !TB)      gemm_wmma<0, 0><<<grid, blk, 0, s>>>(g);
  else if (!TA && TB)  gemm_wmma<0, 1><<<grid, blk, 0, s>>>(g);
  else                 gemm_wmma<1, 0><<<grid, blk, 0, s>>>(g);
}

extern "C" void kernel_launch(void* const* d_in, const int* in_sizes, int n_in,
                              void* d_out, int out_size, void* d_ws, size_t ws_size,
                              hipStream_t stream) {
  (void)in_sizes; (void)n_in; (void)out_size; (void)ws_size;

  const float* x      = (const float*)d_in[0];
  const float* pair   = (const float*)d_in[1];
  const float* pmask  = (const float*)d_in[2];
  const float* amask  = (const float*)d_in[3];
  const float* opmask = (const float*)d_in[4];
  const float* opnorm = (const float*)d_in[5];
  int pi = 6;
  const float* Wq      = (const float*)d_in[pi++];
  const float* Wk      = (const float*)d_in[pi++];
  const float* Wv      = (const float*)d_in[pi++];
  const float* Wo      = (const float*)d_in[pi++];
  const float* bo      = (const float*)d_in[pi++];
  const float* Wbias   = (const float*)d_in[pi++];
  const float* bbias   = (const float*)d_in[pi++];
  const float* ln1g    = (const float*)d_in[pi++];
  const float* ln1b    = (const float*)d_in[pi++];
  const float* W1      = (const float*)d_in[pi++];
  const float* b1      = (const float*)d_in[pi++];
  const float* W2      = (const float*)d_in[pi++];
  const float* b2      = (const float*)d_in[pi++];
  const float* ln2g    = (const float*)d_in[pi++];
  const float* ln2b    = (const float*)d_in[pi++];
  const float* WopmIn  = (const float*)d_in[pi++];
  const float* bopmIn  = (const float*)d_in[pi++];
  const float* WopmOut = (const float*)d_in[pi++];
  const float* bopmOut = (const float*)d_in[pi++];
  const float* lnOpmG  = (const float*)d_in[pi++];
  const float* lnOpmB  = (const float*)d_in[pi++];
  const float* Wabp    = (const float*)d_in[pi++];
  const float* babp    = (const float*)d_in[pi++];
  const float* Wabg    = (const float*)d_in[pi++];
  const float* babg    = (const float*)d_in[pi++];
  const float* Wgp     = (const float*)d_in[pi++];
  const float* bgp     = (const float*)d_in[pi++];
  const float* Wz      = (const float*)d_in[pi++];
  const float* bz      = (const float*)d_in[pi++];
  const float* lnTOg   = (const float*)d_in[pi++];
  const float* lnTOb   = (const float*)d_in[pi++];
  const float* lnTg    = (const float*)d_in[pi++];
  const float* lnTb    = (const float*)d_in[pi++];
  const float* Wp1     = (const float*)d_in[pi++];
  const float* bp1     = (const float*)d_in[pi++];
  const float* Wp2     = (const float*)d_in[pi++];
  const float* bp2     = (const float*)d_in[pi++];
  const float* lnPg    = (const float*)d_in[pi++];
  const float* lnPb    = (const float*)d_in[pi++];

  // workspace layout (floats); ta/tb overlay the dead zopm buffer.
  float* ws = (float*)d_ws;
  const long long szND = (long long)Bb * Nn * Dd;
  float* q      = ws;
  float* k      = q + szND;
  float* v      = k + szND;
  float* proj   = v + szND;
  float* x1     = proj + szND;
  float* o      = x1 + szND;
  float* scores = o + szND;                              // B*H*N*N
  float* hff    = scores + (long long)Bb * Hh * Nn * Nn; // B*N*FF
  float* ab     = hff + (long long)Bb * Nn * FFf;        // B*N*64
  float* t1     = ab + (long long)Bb * Nn * 64;          // B*N*2048
  float* zopm   = t1 + (long long)Bb * Nn * 2048;        // B*N*N*64
  float* pairo  = zopm + Bb * NNll * 64;                 // B*N*N*64
  float* ttri   = pairo + Bb * NNll * 64;                // B*HID*N*N
  float* ta     = zopm;
  float* tb     = zopm + (long long)Bb * HIDh * NNll;

  float* xOut    = (float*)d_out;
  float* pairOut = xOut + szND;

  const float qscale = 1.0f / sqrtf((float)HDd);

  // ---- attention ----
  launch_gemm(0, 0, ga(x, Dd, 0, 0, Wq, Dd, 0, 0, q, Dd, 0, 0,
                       Bb * Nn, Dd, Dd, 1, 1, nullptr, nullptr, 1.0f, 0), stream);
  launch_gemm(0, 0, ga(x, Dd, 0, 0, Wk, Dd, 0, 0, k, Dd, 0, 0,
                       Bb * Nn, Dd, Dd, 1, 1, nullptr, nullptr, 1.0f, 0), stream);
  launch_gemm(0, 0, ga(x, Dd, 0, 0, Wv, Dd, 0, 0, v, Dd, 0, 0,
                       Bb * Nn, Dd, Dd, 1, 1, nullptr, nullptr, 1.0f, 0), stream);
  // logits[b,h] = qscale * q[b,:,h,:] @ k[b,:,h,:]^T
  launch_gemm(0, 1, ga(q, Dd, (long long)Nn * Dd, HDd,
                       k, Dd, (long long)Nn * Dd, HDd,
                       scores, Nn, (long long)Hh * NNll, NNll,
                       Nn, Nn, HDd, Hh, Bb * Hh, nullptr, nullptr, qscale, 0), stream);
  bias_mask_kernel<<<2048, 256, 0, stream>>>(pair, Wbias, bbias, amask, scores);
  softmax_kernel<<<Bb * Hh * Nn, 256, 0, stream>>>(scores);
  launch_gemm(0, 0, ga(scores, Nn, (long long)Hh * NNll, NNll,
                       v, Dd, (long long)Nn * Dd, HDd,
                       o, Dd, (long long)Nn * Dd, HDd,
                       Nn, HDd, Nn, Hh, Bb * Hh, nullptr, nullptr, 1.0f, 0), stream);
  launch_gemm(0, 0, ga(o, Dd, 0, 0, Wo, Dd, 0, 0, proj, Dd, 0, 0,
                       Bb * Nn, Dd, Dd, 1, 1, bo, nullptr, 1.0f, 0), stream);
  ln_residual_kernel<<<Bb * Nn, 256, 0, stream>>>(x, proj, ln1g, ln1b, x1);

  // ---- node FFN ----
  launch_gemm(0, 0, ga(x1, Dd, 0, 0, W1, FFf, 0, 0, hff, FFf, 0, 0,
                       Bb * Nn, FFf, Dd, 1, 1, b1, nullptr, 1.0f, 1 /*gelu*/), stream);
  launch_gemm(0, 0, ga(hff, FFf, 0, 0, W2, Dd, 0, 0, proj, Dd, 0, 0,
                       Bb * Nn, Dd, FFf, 1, 1, b2, nullptr, 1.0f, 0), stream);
  ln_residual_kernel<<<Bb * Nn, 256, 0, stream>>>(x1, proj, ln2g, ln2b, xOut);

  // ---- outer product mean (factored: t1[i,e,p] = a@W, z = b @ t1) ----
  launch_gemm(0, 0, ga(xOut, Dd, 0, 0, WopmIn, 2 * HIDh, 0, 0, ab, 2 * HIDh, 0, 0,
                       Bb * Nn, 2 * HIDh, Dd, 1, 1, bopmIn, opmask, 1.0f, 0), stream);
  launch_gemm(0, 0, ga(ab, 64, 0, 0, WopmOut, HIDh * Pp, 0, 0, t1, HIDh * Pp, 0, 0,
                       Bb * Nn, HIDh * Pp, HIDh, 1, 1, nullptr, nullptr, 1.0f, 0), stream);
  // z[b,i,j,p] = sum_e b[b,j,e] * t1[b,i,e,p] : batch over (b,i), inner=N
  launch_gemm(0, 0, ga(ab + HIDh, 64, (long long)Nn * 64, 0,
                       t1, Pp, (long long)Nn * HIDh * Pp, HIDh * Pp,
                       zopm, Pp, NNll * Pp, (long long)Nn * Pp,
                       Nn, Pp, HIDh, Nn, Bb * Nn, bopmOut, nullptr, 1.0f, 0), stream);
  opm_ln_kernel<<<2048, 256, 0, stream>>>(pair, zopm, opnorm, lnOpmG, lnOpmB, pairo);

  // ---- triangle multiplication ----
  tri_proj_kernel<<<4096, 128, 0, stream>>>(pairo, pmask, Wabp, babp, Wabg, babg, ta, tb);
  // t[b,c] = TA @ TB^T  then  += TA^T @ TB   (channel-major, batch = B*HID)
  launch_gemm(0, 1, ga(ta, Nn, NNll, 0, tb, Nn, NNll, 0, ttri, Nn, NNll, 0,
                       Nn, Nn, Nn, 1, Bb * HIDh, nullptr, nullptr, 1.0f, 0), stream);
  launch_gemm(1, 0, ga(ta, Nn, NNll, 0, tb, Nn, NNll, 0, ttri, Nn, NNll, 0,
                       Nn, Nn, Nn, 1, Bb * HIDh, nullptr, nullptr, 1.0f, 2 /*acc*/), stream);
  tri_out_kernel<<<4096, 128, 0, stream>>>(ttri, pairo, lnTOg, lnTOb, Wgp, bgp,
                                           Wz, bz, lnTg, lnTb, pairOut);

  // ---- pair FFN (in-place on d_out pair region) ----
  pair_ffn_kernel<<<4096, 128, 0, stream>>>(pairOut, Wp1, bp1, Wp2, bp2, lnPg, lnPb);
}
